// SelfAttention_52458730553930
// MI455X (gfx1250) — matrix operand order repaired and checked by reading, not compiled, not run on hardware
//
#include <hip/hip_runtime.h>
#include <stdint.h>

// ---------------------------------------------------------------------------
// Self-attention (N=4, C=512, HW=4096, OC=64), flash-attention style, bf16
// WMMA + f32 accumulation.
//  - weights pre-converted to bf16 once; x-tile converted at LDS staging
//  - V stored in WMMA-B fragment-native 32x16 tiles; a j-step's 32 tiles are
//    one contiguous 32KB panel
//  - attention: waves 0/1 produce P (online softmax) for two i-tiles; wave 2
//    drives a Tensor-Data-Mover pipeline that prefetches the next V panel to
//    LDS (triple-buffered, s_wait_tensorcnt); all 8 waves consume P x V
// ---------------------------------------------------------------------------

typedef __attribute__((ext_vector_type(16))) __bf16 v16bf;
typedef __attribute__((ext_vector_type(8)))  float  v8f;
typedef __attribute__((ext_vector_type(4)))  unsigned v4u;
typedef __attribute__((ext_vector_type(8)))  int      v8i;
typedef __attribute__((ext_vector_type(4)))  int      v4i;

#define NB     4
#define CDIM   512
#define HWDIM  4096
#define OCDIM  64
#define ITILES 2
#define PANEL_DW (32 * 256)   // one j-panel: 32 tiles x 256 dwords = 32KB
#define XS_US  520
#define XS_U   260

static __device__ __forceinline__ unsigned short f2bf(float f) {
  union { float f; unsigned u; } x; x.f = f;
  unsigned r = x.u + 0x7FFFu + ((x.u >> 16) & 1u);   // round-to-nearest-even
  return (unsigned short)(r >> 16);
}
static __device__ __forceinline__ unsigned packbf(float lo, float hi) {
  return (unsigned)f2bf(lo) | ((unsigned)f2bf(hi) << 16);
}

union Frag { v16bf v; unsigned u[8]; unsigned short h[16]; };

// 16-bit A-matrix (16x32), contiguous k order used here: dword p of a lane holds the K pair starting
// at  hi*16+2p   (returned as dword index)
static __device__ __forceinline__ int a_keh(int p, int hi) {
  return hi * 8 + p;
}
// 16-bit B-matrix (32x16), same contiguous k order: lane (n, hi); dword p holds K pair hi*16+2p.

// ---------------------------------------------------------------------------
// TDM: issue an async 32KB (8192-dword) 1-D tile copy, global -> LDS.
// D# built per CDNA5 ISA 8.3/8.4: group0 = {count=1 | lds_addr | global_addr |
// type=2}, group1 = {data_size=4B, tensor_dim0, tensor_dim1=1, tile_dim0}.
// This toolchain exposes the 6-arg builtin (extra int32x8 group; zero-filled
// like the other unused descriptor groups).
// ---------------------------------------------------------------------------
static __device__ __forceinline__ void tdm_load_panel(unsigned lds_byte_off,
                                                      const void* gsrc) {
  unsigned long long ga = (unsigned long long)(uintptr_t)gsrc;
  v4u g0;
  g0.x = 1u;                                            // count=1 (valid)
  g0.y = lds_byte_off;                                  // lds_addr (bytes)
  g0.z = (unsigned)ga;                                  // global_addr[31:0]
  g0.w = (unsigned)((ga >> 32) & 0x01FFFFFFu) | (2u << 30);  // [56:32] | type=2
  const unsigned nd = PANEL_DW;                         // 8192 dwords
  v8i g1;
  g1[0] = (int)(2u << 16);                              // data_size = 4 bytes
  g1[1] = (int)((nd & 0xFFFFu) << 16);                  // tensor_dim0[15:0]
  g1[2] = (int)((nd >> 16) | (1u << 16));               // dim0[31:16] | dim1=1
  g1[3] = (int)((nd & 0xFFFFu) << 16);                  // tile_dim0 = 8192
  g1[4] = 0;                                            // tile_dim1/2 unused
  g1[5] = (int)nd;                                      // tensor_dim0_stride
  g1[6] = 0;
  g1[7] = 0;
  v4i z4 = {0, 0, 0, 0};
  v8i z8 = {0, 0, 0, 0, 0, 0, 0, 0};
  __builtin_amdgcn_tensor_load_to_lds(g0, g1, z4, z4, z8, 0);
}

// ---------------------------------------------------------------------------
// Kernel 0: f32 -> packed bf16 conversion (weights; off the hot path)
// ---------------------------------------------------------------------------
__global__ __launch_bounds__(256) void cvt_bf16_kernel(
    const float* __restrict__ src, unsigned* __restrict__ dst, int npairs) {
  int i = blockIdx.x * 256 + threadIdx.x;
  if (i < npairs) {
    float2 v = *(const float2*)(src + 2 * (size_t)i);
    dst[i] = packbf(v.x, v.y);
  }
}

// ---------------------------------------------------------------------------
// Kernel 1: QKV projections.  grid=(HW/16, N), block=256 (8 waves).
// ---------------------------------------------------------------------------
static __device__ __forceinline__ v8f proj_mm(
    const unsigned* __restrict__ xsu,
    const unsigned short* __restrict__ wbf,
    const float* __restrict__ bm, int obase, int nl, int hi) {
  float bias = bm[obase + nl];
  v8f acc;
  #pragma unroll
  for (int r = 0; r < 8; ++r) acc[r] = bias;
  const unsigned* wu = (const unsigned*)wbf + (size_t)(obase + nl) * (CDIM / 2);
  for (int cb = 0; cb < CDIM; cb += 32) {
    Frag a, b;
    #pragma unroll
    for (int p = 0; p < 8; ++p) {
      a.u[p] = xsu[nl * XS_U + (cb >> 1) + a_keh(p, hi)];
      b.u[p] = wu[(cb >> 1) + hi * 8 + p];
    }
    acc = __builtin_amdgcn_wmma_f32_16x16x32_bf16(false, a.v, false, b.v,
                                                  (short)0, acc, false, false);
  }
  return acc;
}

__global__ __launch_bounds__(256) void qkv_proj_kernel(
    const float* __restrict__ x,
    const float* __restrict__ bk, const float* __restrict__ bq,
    const float* __restrict__ bv,
    const unsigned short* __restrict__ wkb, const unsigned short* __restrict__ wqb,
    const unsigned short* __restrict__ wvb,
    unsigned short* __restrict__ qws, unsigned short* __restrict__ kws,
    unsigned* __restrict__ vtiles) {
  __shared__ unsigned short xsb[16 * XS_US];   // bf16 x-tile: xsb[i_local][c]

  const int n   = blockIdx.y;
  const int i0  = blockIdx.x * 16;
  const int tid = threadIdx.x;
  const float* xb = x + (size_t)n * CDIM * HWDIM;
  unsigned* xr = (unsigned*)xsb;

  for (int idx = tid; idx < 1024; idx += 256) {
    int c2 = idx >> 2;
    int ii = (idx & 3) << 2;
    const float4 v0 = *(const float4*)(xb + (size_t)(2 * c2)     * HWDIM + i0 + ii);
    const float4 v1 = *(const float4*)(xb + (size_t)(2 * c2 + 1) * HWDIM + i0 + ii);
    xr[(ii + 0) * XS_U + c2] = packbf(v0.x, v1.x);
    xr[(ii + 1) * XS_U + c2] = packbf(v0.y, v1.y);
    xr[(ii + 2) * XS_U + c2] = packbf(v0.z, v1.z);
    xr[(ii + 3) * XS_U + c2] = packbf(v0.w, v1.w);
  }
  __syncthreads();

  const int wave = tid >> 5;
  const int L  = tid & 31;
  const int nl = L & 15;
  const int hi = L >> 4;
  const unsigned* xsu = (const unsigned*)xsb;

  for (int s = 0; s < 5; ++s) {
    int t = wave * 5 + s;
    if (t < 8) {   // Q or K: row-major (i, 64) bf16
      int obase = (t & 3) * 16;
      const unsigned short* w = (t < 4) ? wqb : wkb;
      const float* b = (t < 4) ? bq : bk;
      unsigned short* dst = ((t < 4) ? qws : kws) + (size_t)n * HWDIM * OCDIM;
      v8f acc = proj_mm(xsu, w, b, obase, nl, hi);
      #pragma unroll
      for (int r = 0; r < 8; ++r) {
        int row = r + (hi << 3);
        dst[(size_t)(i0 + row) * OCDIM + obase + nl] = f2bf(acc[r]);
      }
    } else {       // V: fragment-native 32x16 tiles, one b128 store per lane
      int obase = (t - 8) * 16;
      v8f acc = proj_mm(xsu, wvb, bv, obase, nl, hi);
      const int jt   = i0 >> 5;
      const int half = (i0 >> 4) & 1;
      unsigned* du = vtiles + (size_t)n * (HWDIM * CDIM / 2)
                   + ((size_t)jt * (CDIM / 16) + (obase >> 4)) * 256
                   + (size_t)(nl + 16 * half) * 8 + 4 * hi;
      uint4 val;
      val.x = packbf(acc[0], acc[1]);
      val.y = packbf(acc[2], acc[3]);
      val.z = packbf(acc[4], acc[5]);
      val.w = packbf(acc[6], acc[7]);
      *(uint4*)du = val;
    }
  }
}

// ---------------------------------------------------------------------------
// Kernel 2: flash attention + residual.  grid=(HW/32, N), block=256.
// ---------------------------------------------------------------------------
__global__ __launch_bounds__(256) void attn_kernel(
    const float* __restrict__ x, const float* __restrict__ gamma,
    const unsigned short* __restrict__ qws, const unsigned short* __restrict__ kws,
    const unsigned* __restrict__ vtiles, float* __restrict__ out) {
  __shared__ __align__(16) unsigned vpanel[3][PANEL_DW];   // 3 x 32KB V panels
  __shared__ unsigned Pbuf[2][ITILES][16 * 16];
  __shared__ float alphas[2][ITILES][16];
  __shared__ float lbuf[ITILES][16];

  const int n   = blockIdx.y;
  const int ib  = blockIdx.x * (16 * ITILES);
  const int tid = threadIdx.x;
  const int wave = tid >> 5;
  const int L  = tid & 31;
  const int nl = L & 15;
  const int hi = L >> 4;

  const unsigned short* qb = qws + (size_t)n * HWDIM * OCDIM;
  const unsigned short* kb = kws + (size_t)n * HWDIM * OCDIM;
  const unsigned* vtb = vtiles + (size_t)n * (HWDIM * CDIM / 2);

  // wave 2: kick off the TDM pipeline with the first V panel
  if (wave == 2)
    tdm_load_panel((unsigned)(uintptr_t)&vpanel[0][0], vtb);

  Frag qa[2];
  if (wave < ITILES) {
    const int i0 = ib + wave * 16;
    #pragma unroll
    for (int t = 0; t < 2; ++t) {
      const unsigned* qru = (const unsigned*)(qb + (size_t)(i0 + nl) * OCDIM + t * 32);
      #pragma unroll
      for (int p = 0; p < 8; ++p) qa[t].u[p] = qru[a_keh(p, hi)];
    }
  }

  v8f acc[ITILES][4];
  #pragma unroll
  for (int it = 0; it < ITILES; ++it)
    #pragma unroll
    for (int k = 0; k < 4; ++k)
      #pragma unroll
      for (int r = 0; r < 8; ++r) acc[it][k][r] = 0.0f;

  float m_st[8], l_st[8];
  #pragma unroll
  for (int r = 0; r < 8; ++r) { m_st[r] = -1e30f; l_st[r] = 0.0f; }

  for (int step = 0; step < HWDIM / 32; ++step) {
    const int j0 = step * 32;
    const int pb = step & 1;
    const int vb = step % 3;

    if (wave < ITILES) {
      v8f s0, s1;
      #pragma unroll
      for (int r = 0; r < 8; ++r) { s0[r] = 0.0f; s1[r] = 0.0f; }
      #pragma unroll
      for (int t = 0; t < 2; ++t) {   // contraction over oc = 64
        Frag b0, b1;
        const unsigned* kr0 = (const unsigned*)(kb + (size_t)(j0 + nl) * OCDIM + t * 32);
        const unsigned* kr1 = (const unsigned*)(kb + (size_t)(j0 + 16 + nl) * OCDIM + t * 32);
        #pragma unroll
        for (int p = 0; p < 8; ++p) {
          b0.u[p] = kr0[hi * 8 + p];
          b1.u[p] = kr1[hi * 8 + p];
        }
        s0 = __builtin_amdgcn_wmma_f32_16x16x32_bf16(false, qa[t].v, false, b0.v,
                                                     (short)0, s0, false, false);
        s1 = __builtin_amdgcn_wmma_f32_16x16x32_bf16(false, qa[t].v, false, b1.v,
                                                     (short)0, s1, false, false);
      }
      #pragma unroll
      for (int r = 0; r < 8; ++r) {
        float mx = fmaxf(s0[r], s1[r]);
        mx = fmaxf(mx, __shfl_xor(mx, 1, 16));
        mx = fmaxf(mx, __shfl_xor(mx, 2, 16));
        mx = fmaxf(mx, __shfl_xor(mx, 4, 16));
        mx = fmaxf(mx, __shfl_xor(mx, 8, 16));
        float mnew = fmaxf(m_st[r], mx);
        float al = __expf(m_st[r] - mnew);
        float p0 = __expf(s0[r] - mnew);
        float p1 = __expf(s1[r] - mnew);
        float sum = p0 + p1;
        sum += __shfl_xor(sum, 1, 16);
        sum += __shfl_xor(sum, 2, 16);
        sum += __shfl_xor(sum, 4, 16);
        sum += __shfl_xor(sum, 8, 16);
        l_st[r] = l_st[r] * al + sum;
        m_st[r] = mnew;
        int row = r + (hi << 3);
        if (nl == 0) alphas[pb][wave][row] = al;
        unsigned short* pr = (unsigned short*)&Pbuf[pb][wave][row * 16];
        pr[nl]      = f2bf(p0);
        pr[16 + nl] = f2bf(p1);
      }
    }

    // wave 2: prefetch next panel, then ensure the current one has landed.
    // (tensor ops complete in order per wave -> tensorcnt<=1 frees the older)
    if (wave == 2) {
      if (step + 1 < HWDIM / 32) {
        tdm_load_panel((unsigned)(uintptr_t)&vpanel[(step + 1) % 3][0],
                       vtb + (size_t)(step + 1) * PANEL_DW);
        __builtin_amdgcn_s_wait_tensorcnt(1);
      } else {
        __builtin_amdgcn_s_wait_tensorcnt(0);
      }
    }
    __syncthreads();   // one barrier/step: P double-buffered, V triple-buffered

    Frag pa[ITILES];
    float alr[ITILES][8];
    #pragma unroll
    for (int it = 0; it < ITILES; ++it) {
      #pragma unroll
      for (int p = 0; p < 8; ++p)
        pa[it].u[p] = Pbuf[pb][it][nl * 16 + a_keh(p, hi)];
      #pragma unroll
      for (int r = 0; r < 8; ++r) alr[it][r] = alphas[pb][it][r + (hi << 3)];
    }
    #pragma unroll
    for (int k = 0; k < 4; ++k) {
      const int ct = wave * 4 + k;
      const uint4* vp = (const uint4*)&vpanel[vb][ct * 256 + L * 8];
      uint4 vlo = vp[0];
      uint4 vhi = vp[1];
      Frag vfr;
      vfr.u[0] = vlo.x; vfr.u[1] = vlo.y; vfr.u[2] = vlo.z; vfr.u[3] = vlo.w;
      vfr.u[4] = vhi.x; vfr.u[5] = vhi.y; vfr.u[6] = vhi.z; vfr.u[7] = vhi.w;
      #pragma unroll
      for (int it = 0; it < ITILES; ++it) {
        v8f a2 = acc[it][k];
        #pragma unroll
        for (int r = 0; r < 8; ++r) a2[r] *= alr[it][r];
        acc[it][k] = __builtin_amdgcn_wmma_f32_16x16x32_bf16(
            false, pa[it].v, false, vfr.v, (short)0, a2, false, false);
      }
    }
  }

  __syncthreads();
  if (wave < ITILES && nl == 0) {
    #pragma unroll
    for (int r = 0; r < 8; ++r) lbuf[wave][r + (hi << 3)] = l_st[r];
  }
  __syncthreads();

  const float g = gamma[0];
  const float* xb = x   + (size_t)n * CDIM * HWDIM;
  float*       ob = out + (size_t)n * CDIM * HWDIM;
  #pragma unroll
  for (int it = 0; it < ITILES; ++it) {
    float inv[8];
    #pragma unroll
    for (int r = 0; r < 8; ++r) inv[r] = 1.0f / lbuf[it][r + (hi << 3)];
    #pragma unroll
    for (int k = 0; k < 4; ++k) {
      int c = wave * 64 + k * 16 + nl;
      #pragma unroll
      for (int r = 0; r < 8; ++r) {
        int row = r + (hi << 3);
        size_t idx = (size_t)c * HWDIM + (ib + it * 16 + row);
        ob[idx] = xb[idx] + g * acc[it][k][r] * inv[r];
      }
    }
  }
}

// ---------------------------------------------------------------------------
extern "C" void kernel_launch(void* const* d_in, const int* in_sizes, int n_in,
                              void* d_out, int out_size, void* d_ws, size_t ws_size,
                              hipStream_t stream) {
  const float* x     = (const float*)d_in[0];
  const float* Wk    = (const float*)d_in[1];
  const float* bk    = (const float*)d_in[2];
  const float* Wq    = (const float*)d_in[3];
  const float* bq    = (const float*)d_in[4];
  const float* Wv    = (const float*)d_in[5];
  const float* bv    = (const float*)d_in[6];
  const float* gamma = (const float*)d_in[7];
  float* out = (float*)d_out;

  unsigned short* qws = (unsigned short*)d_ws;
  unsigned short* kws = qws + (size_t)NB * HWDIM * OCDIM;
  unsigned short* vws = kws + (size_t)NB * HWDIM * OCDIM;
  unsigned short* wkb = vws + (size_t)NB * HWDIM * CDIM;
  unsigned short* wqb = wkb + (size_t)OCDIM * CDIM;
  unsigned short* wvb = wqb + (size_t)OCDIM * CDIM;

  const int pk = OCDIM * CDIM / 2;
  const int pv = CDIM * CDIM / 2;
  cvt_bf16_kernel<<<(pk + 255) / 256, 256, 0, stream>>>(Wk, (unsigned*)wkb, pk);
  cvt_bf16_kernel<<<(pk + 255) / 256, 256, 0, stream>>>(Wq, (unsigned*)wqb, pk);
  cvt_bf16_kernel<<<(pv + 255) / 256, 256, 0, stream>>>(Wv, (unsigned*)wvb, pv);

  dim3 grid1(HWDIM / 16, NB);
  qkv_proj_kernel<<<grid1, 256, 0, stream>>>(x, bk, bq, bv, wkb, wqb, wvb,
                                             qws, kws, (unsigned*)vws);
  dim3 grid2(HWDIM / (16 * ITILES), NB);
  attn_kernel<<<grid2, 256, 0, stream>>>(x, gamma, qws, kws, (unsigned*)vws, out);
}
